// PatchTSTRopeAttention_75299366633549
// MI455X (gfx1250) — matrix operand order repaired and checked by reading, not compiled, not run on hardware
//
#include <hip/hip_runtime.h>
#include <hip/hip_bf16.h>

typedef __attribute__((ext_vector_type(16))) _Float16 v16h;
typedef __attribute__((ext_vector_type(8)))  float    v8f;

#define D_MODEL 768
#define NHEADS  12
#define HD      64
#define BATCH   16
#define SEQ     1024
#define NROWS   (BATCH * SEQ)           /* 16384 */
#define NX      (NROWS * D_MODEL)       /* 12582912 */
#define NW      (D_MODEL * D_MODEL)     /* 589824  */
#define OFF_WQ  (NX)
#define OFF_WK  (NX + NW)
#define OFF_WV  (NX + 2 * NW)
#define OFF_WO  (NX + 3 * NW)
#define OFF_Q   (NX + 4 * NW)           /* 14942208 */
#define NQ      (BATCH * NHEADS * SEQ * HD) /* 12582912 */
#define OFF_K   (OFF_Q + NQ)
#define OFF_VT  (OFF_K + NQ)
#define OFF_AO  (OFF_VT + NQ)
#define SCALING 0.125f                  /* 64^-0.5 */
#define LN1E4_OVER_32 0.28782313662425572f

__device__ __forceinline__ v8f wmma_f16(v16h a, v16h b, v8f c) {
  return __builtin_amdgcn_wmma_f32_16x16x32_f16(false, a, false, b,
                                                (short)0, c, false, false);
}

// Build a v16h fragment from two 16-byte runs (addresses 16B aligned).
__device__ __forceinline__ v16h load_frag(const _Float16* p0, const _Float16* p1) {
  v16h r;
  uint4* rp = reinterpret_cast<uint4*>(&r);
  rp[0] = *reinterpret_cast<const uint4*>(p0);
  rp[1] = *reinterpret_cast<const uint4*>(p1);
  return r;
}

// Async global->LDS copy of 16 bytes per lane (CDNA5, tracked by ASYNCcnt).
__device__ __forceinline__ void async_copy_b128(const _Float16* g, const _Float16* l) {
  unsigned lds_off = (unsigned)(size_t)(const void*)l;   // low 32 bits = LDS offset
  unsigned long long ga = (unsigned long long)(size_t)(const void*)g;
  asm volatile("global_load_async_to_lds_b128 %0, %1, off"
               :: "v"(lds_off), "v"(ga) : "memory");
}

// ---------------------------------------------------------------- prep: f32->f16
__global__ void prep_convert(const float* __restrict__ hs,
                             const float* __restrict__ Wq,
                             const float* __restrict__ Wk,
                             const float* __restrict__ Wv,
                             const float* __restrict__ Wo,
                             _Float16* __restrict__ ws) {
  for (int i = blockIdx.x * blockDim.x + threadIdx.x; i < OFF_Q;
       i += gridDim.x * blockDim.x) {
    float v;
    if (i < NX) {
      v = hs[i];
    } else {
      int j = i - NX;
      int w = j / NW, r = j % NW;
      v = (w == 0 ? Wq : w == 1 ? Wk : w == 2 ? Wv : Wo)[r];
    }
    ws[i] = (_Float16)v;
  }
}

// ------------------------------------------------- QKV projection + bias + p-RoPE
// grid: 3 (mat) * 12 (head) * 128 (row groups), 4 waves/block, one 32x64 tile/wave
__global__ __launch_bounds__(128) void qkv_rope(const _Float16* __restrict__ ws,
                                                const float* __restrict__ bq,
                                                const float* __restrict__ bk,
                                                const float* __restrict__ bv,
                                                _Float16* wsw) {
  const int lane = threadIdx.x & 31;
  const int wave = threadIdx.x >> 5;
  const int ln   = lane & 15;
  const bool hi  = lane >= 16;

  const int bi  = blockIdx.x;
  const int mat = bi / (NHEADS * 128);
  const int rem = bi % (NHEADS * 128);
  const int h   = rem / 128;
  const int rg  = rem % 128;
  const int m0  = (rg * 4 + wave) * 32;   // global row base (over B*T)
  const int c0  = h * HD;                 // column base

  const _Float16* Xh = ws;
  const _Float16* Wh = ws + OFF_WQ + mat * NW;
  const float* bias  = (mat == 0) ? bq : (mat == 1) ? bk : bv;

  v8f acc[2][4] = {};

  const _Float16* arow0 = Xh + (m0 + ln) * D_MODEL + (hi ? 8 : 0);
  const _Float16* arow1 = arow0 + 16 * D_MODEL;
  for (int kk = 0; kk < D_MODEL / 32; ++kk) {
    const int k0 = kk * 32;
    v16h a0 = load_frag(arow0 + k0, arow0 + k0 + 16);
    v16h a1 = load_frag(arow1 + k0, arow1 + k0 + 16);
#pragma unroll
    for (int t = 0; t < 4; ++t) {
      const _Float16* brow = Wh + (c0 + t * 16 + ln) * D_MODEL + k0 + (hi ? 8 : 0);
      v16h b = load_frag(brow, brow + 16);
      acc[0][t] = wmma_f16(a0, b, acc[0][t]);
      acc[1][t] = wmma_f16(a1, b, acc[1][t]);
    }
  }

  // bias (column-dependent only)
#pragma unroll
  for (int t = 0; t < 4; ++t) {
    float bb = bias[c0 + t * 16 + ln];
#pragma unroll
    for (int r = 0; r < 2; ++r)
#pragma unroll
      for (int j = 0; j < 8; ++j) acc[r][t][j] += bb;
  }

  // p-RoPE: pair (d, d+32) lives in acc tiles (0,2); pairs 16..31 are identity
  if (mat < 2) {
    const float inv_ts = __expf(-(float)ln * LN1E4_OVER_32);
#pragma unroll
    for (int r = 0; r < 2; ++r)
#pragma unroll
      for (int j = 0; j < 8; ++j) {
        float pos = (float)((m0 + r * 16 + j + (hi ? 8 : 0)) & (SEQ - 1));
        float ang = pos * inv_ts;
        float sn, cs;
        __sincosf(ang, &sn, &cs);
        float x1 = acc[r][0][j], x2 = acc[r][2][j];
        acc[r][0][j] = x1 * cs - x2 * sn;
        acc[r][2][j] = x2 * cs + x1 * sn;
      }
    if (mat == 0) {
#pragma unroll
      for (int r = 0; r < 2; ++r)
#pragma unroll
        for (int t = 0; t < 4; ++t)
#pragma unroll
          for (int j = 0; j < 8; ++j) acc[r][t][j] *= SCALING;
    }
  }

  const int b  = m0 / SEQ;
  const int bh = b * NHEADS + h;
  const int sb = (m0 & (SEQ - 1));
  if (mat == 2) {
    _Float16* Vt = wsw + OFF_VT;   // [bh, d, s]
#pragma unroll
    for (int r = 0; r < 2; ++r)
#pragma unroll
      for (int t = 0; t < 4; ++t)
#pragma unroll
        for (int j = 0; j < 8; ++j) {
          int s = sb + r * 16 + j + (hi ? 8 : 0);
          Vt[(bh * HD + t * 16 + ln) * SEQ + s] = (_Float16)acc[r][t][j];
        }
  } else {
    _Float16* Out = wsw + (mat == 0 ? OFF_Q : OFF_K);  // [bh, s, d]
#pragma unroll
    for (int r = 0; r < 2; ++r)
#pragma unroll
      for (int t = 0; t < 4; ++t)
#pragma unroll
        for (int j = 0; j < 8; ++j) {
          int s = sb + r * 16 + j + (hi ? 8 : 0);
          Out[(bh * SEQ + s) * HD + t * 16 + ln] = (_Float16)acc[r][t][j];
        }
  }
}

// ------------------------------------------------------ flash attention
// 1 wave = 16 query rows x full head. All 4 waves of a block share (b,h), so K/V
// 32-key tiles are staged into LDS with async copies (double buffered) and shared.
__global__ __launch_bounds__(128) void attn(const _Float16* __restrict__ ws,
                                            _Float16* wsw) {
  __shared__ __align__(16) _Float16 kbuf[2][32 * 64];  // [key][d]
  __shared__ __align__(16) _Float16 vbuf[2][64 * 32];  // [d][key]
  __shared__ __align__(16) _Float16 pbuf[4][16 * 32];  // per-wave P tile

  const int lane = threadIdx.x & 31;
  const int wave = threadIdx.x >> 5;
  const int ln   = lane & 15;
  const bool hi  = lane >= 16;

  const int qt = blockIdx.x * 4 + wave;
  const int bh = qt >> 6;
  const int m0 = (qt & 63) * 16;

  const _Float16* Qh = ws + OFF_Q + bh * SEQ * HD;
  const _Float16* Kh = ws + OFF_K + bh * SEQ * HD;
  const _Float16* Vt = ws + OFF_VT + bh * HD * SEQ;
  _Float16* pw = pbuf[wave];

  // Issue async global->LDS copies for one 32-key K tile + V tile (4KB each).
  auto issue_tile = [&](int kt, int sel) {
    const int key0 = kt * 32;
#pragma unroll
    for (int it = 0; it < 2; ++it) {
      int c = (wave * 2 + it) * 32 + lane;          // 0..255 16B chunks
      {
        int key = c >> 3, off = (c & 7) * 8;        // K: 32 rows x 128B
        async_copy_b128(Kh + (key0 + key) * HD + off, &kbuf[sel][key * 64 + off]);
      }
      {
        int row = c >> 2, off = (c & 3) * 8;        // V: 64 rows x 64B
        async_copy_b128(Vt + row * SEQ + key0 + off, &vbuf[sel][row * 32 + off]);
      }
    }
  };

  const _Float16* qrow = Qh + (m0 + ln) * HD + (hi ? 8 : 0);
  v16h aq0 = load_frag(qrow, qrow + 16);          // d 0..31
  v16h aq1 = load_frag(qrow + 32, qrow + 48);     // d 32..63

  float M[8], L[8];
#pragma unroll
  for (int j = 0; j < 8; ++j) { M[j] = -3.0e38f; L[j] = 0.0f; }
  v8f o[4] = {};

  issue_tile(0, 0);
  for (int kt = 0; kt < SEQ / 32; ++kt) {
    const int sel = kt & 1;
    asm volatile("s_wait_asynccnt 0" ::: "memory");   // this wave's tile ready
    __syncthreads();                                  // whole block's tile ready
    if (kt + 1 < SEQ / 32) issue_tile(kt + 1, sel ^ 1);

    // scores: S = (Q*scale) . K^T for 32 keys, from LDS
    v8f s0 = {}, s1 = {};
    {
      const _Float16* kr = &kbuf[sel][ln * 64 + (hi ? 8 : 0)];
      v16h b0 = load_frag(kr, kr + 16);           // d 0..31
      v16h b1 = load_frag(kr + 32, kr + 48);      // d 32..63
      s0 = wmma_f16(aq0, b0, s0);
      s0 = wmma_f16(aq1, b1, s0);
      const _Float16* kr2 = kr + 16 * 64;         // keys 16..31
      v16h b2 = load_frag(kr2, kr2 + 16);
      v16h b3 = load_frag(kr2 + 32, kr2 + 48);
      s1 = wmma_f16(aq0, b2, s1);
      s1 = wmma_f16(aq1, b3, s1);
    }

    // online softmax; row j+8*hi spans one 16-lane half in the C layout
#pragma unroll
    for (int j = 0; j < 8; ++j) {
      float a0 = s0[j], a1 = s1[j];
      float mx = fmaxf(a0, a1);
      for (int d = 1; d < 16; d <<= 1) mx = fmaxf(mx, __shfl_xor(mx, d, 16));
      float newM = fmaxf(M[j], mx);
      float scale = __expf(M[j] - newM);
      M[j] = newM;
      float p0 = __expf(a0 - newM);
      float p1 = __expf(a1 - newM);
      float rs = p0 + p1;
      for (int d = 1; d < 16; d <<= 1) rs += __shfl_xor(rs, d, 16);
      L[j] = L[j] * scale + rs;
      o[0][j] *= scale; o[1][j] *= scale; o[2][j] *= scale; o[3][j] *= scale;
      int ml = j + (hi ? 8 : 0);
      pw[ml * 32 + ln]      = (_Float16)p0;
      pw[ml * 32 + 16 + ln] = (_Float16)p1;
    }
    asm volatile("s_wait_dscnt 0" ::: "memory");

    // P (16x32, A layout) from LDS; V^T tile in LDS gives contiguous B fragments
    const _Float16* pr = pw + ln * 32 + (hi ? 8 : 0);
    v16h ap = load_frag(pr, pr + 16);
#pragma unroll
    for (int t = 0; t < 4; ++t) {
      const _Float16* vr = &vbuf[sel][(t * 16 + ln) * 32 + (hi ? 8 : 0)];
      v16h bv = load_frag(vr, vr + 16);
      o[t] = wmma_f16(ap, bv, o[t]);
    }
  }

  const int b = bh / NHEADS, h = bh % NHEADS;
  _Float16* AO = wsw + OFF_AO;   // [b, s, h, d] -> rows of 768 for out proj
#pragma unroll
  for (int t = 0; t < 4; ++t)
#pragma unroll
    for (int j = 0; j < 8; ++j) {
      int s = m0 + j + (hi ? 8 : 0);
      AO[((b * SEQ + s) * NHEADS + h) * HD + t * 16 + ln] =
          (_Float16)(o[t][j] * (1.0f / L[j]));
    }
}

// ------------------------------------------------------------- output projection
// grid: 12 (col groups) * 128 (row groups), 4 waves/block, one 32x64 tile/wave
__global__ __launch_bounds__(128) void out_proj(const _Float16* __restrict__ ws,
                                                const float* __restrict__ bo,
                                                float* __restrict__ out) {
  const int lane = threadIdx.x & 31;
  const int wave = threadIdx.x >> 5;
  const int ln   = lane & 15;
  const bool hi  = lane >= 16;

  const int bi = blockIdx.x;
  const int cg = bi / 128;            // column group (head-sized, 64 cols)
  const int rg = bi % 128;
  const int m0 = (rg * 4 + wave) * 32;
  const int c0 = cg * 64;

  const _Float16* A = ws + OFF_AO;
  const _Float16* W = ws + OFF_WO;

  v8f acc[2][4] = {};
  const _Float16* arow0 = A + (m0 + ln) * D_MODEL + (hi ? 8 : 0);
  const _Float16* arow1 = arow0 + 16 * D_MODEL;
  for (int kk = 0; kk < D_MODEL / 32; ++kk) {
    const int k0 = kk * 32;
    v16h a0 = load_frag(arow0 + k0, arow0 + k0 + 16);
    v16h a1 = load_frag(arow1 + k0, arow1 + k0 + 16);
#pragma unroll
    for (int t = 0; t < 4; ++t) {
      const _Float16* brow = W + (c0 + t * 16 + ln) * D_MODEL + k0 + (hi ? 8 : 0);
      v16h b = load_frag(brow, brow + 16);
      acc[0][t] = wmma_f16(a0, b, acc[0][t]);
      acc[1][t] = wmma_f16(a1, b, acc[1][t]);
    }
  }
#pragma unroll
  for (int t = 0; t < 4; ++t) {
    float bb = bo[c0 + t * 16 + ln];
#pragma unroll
    for (int r = 0; r < 2; ++r)
#pragma unroll
      for (int j = 0; j < 8; ++j) {
        int m = m0 + r * 16 + j + (hi ? 8 : 0);
        out[m * D_MODEL + c0 + t * 16 + ln] = acc[r][t][j] + bb;
      }
  }
}

// -------------------------------------------------------------------- launcher
extern "C" void kernel_launch(void* const* d_in, const int* in_sizes, int n_in,
                              void* d_out, int out_size, void* d_ws, size_t ws_size,
                              hipStream_t stream) {
  const float* hs = (const float*)d_in[0];
  const float* Wq = (const float*)d_in[1];
  const float* bq = (const float*)d_in[2];
  const float* Wk = (const float*)d_in[3];
  const float* bk = (const float*)d_in[4];
  const float* Wv = (const float*)d_in[5];
  const float* bv = (const float*)d_in[6];
  const float* Wo = (const float*)d_in[7];
  const float* bo = (const float*)d_in[8];
  _Float16* ws = (_Float16*)d_ws;
  float* out = (float*)d_out;

  prep_convert<<<8192, 256, 0, stream>>>(hs, Wq, Wk, Wv, Wo, ws);
  qkv_rope<<<3 * NHEADS * 128, 128, 0, stream>>>(ws, bq, bk, bv, ws);
  attn<<<(BATCH * NHEADS * 64) / 4, 128, 0, stream>>>(ws, ws);
  out_proj<<<NHEADS * 128, 128, 0, stream>>>(ws, bo, out);
}
